// DecoderBlock_36352603193653
// MI455X (gfx1250) — matrix operand order repaired and checked by reading, not compiled, not run on hardware
//
#include <hip/hip_runtime.h>
#include <hip/hip_fp16.h>

typedef __attribute__((ext_vector_type(16))) _Float16 v16h;
typedef __attribute__((ext_vector_type(8)))  float    v8f;
typedef unsigned int u32;

union FragAB { v16h h; u32 u[8]; };
union H8 { uint4 q; _Float16 h[8]; };
union H4 { uint2 u; _Float16 h[4]; };

// ---------------------------------------------------------------------------
// WMMA fragment loaders (this file's contiguous k order, same for A and B, wave32)
// A 16x32 f16: lanes0-15 row=lane, K 0..15; lanes16-31 K 16..31
// B 32x16 f16 (stored transposed [N][K] in LDS): lanes0-15 col=lane K0..15,
//             lanes16-31 col=lane-16 K16..31; VGPR j holds K=2j,2j+1
// C/D 16x16 f32: VGPR e -> row = e + 8*(lane>=16), col = lane&15
// ---------------------------------------------------------------------------
__device__ __forceinline__ v16h load_frag_a(const _Float16* p0, int stride, int lane) {
  FragAB f;
  const _Float16* p = p0 + (size_t)(lane & 15) * stride + ((lane >> 4) << 4);
#pragma unroll
  for (int j = 0; j < 4; ++j) {
    f.u[j]     = *(const u32*)(p + 2 * j);
    f.u[4 + j] = *(const u32*)(p + 8 + 2 * j);
  }
  return f.h;
}

__device__ __forceinline__ v16h load_frag_b(const _Float16* p0, int stride, int lane) {
  FragAB f;
  const _Float16* p = p0 + (size_t)(lane & 15) * stride + ((lane >> 4) << 4);
#pragma unroll
  for (int j = 0; j < 8; ++j) f.u[j] = *(const u32*)(p + 2 * j);
  return f.h;
}

// ---------------------------------------------------------------------------
// fp32 -> f16 conversion (weights, emb_motion)
// ---------------------------------------------------------------------------
__global__ __launch_bounds__(256)
void cvt_kernel(const float* __restrict__ x, _Float16* __restrict__ y, int n4) {
  int i = blockIdx.x * 256 + threadIdx.x;
  if (i < n4) {
    float4 v = *(const float4*)(x + (size_t)i * 4);
    H4 o;
    o.h[0] = (_Float16)v.x; o.h[1] = (_Float16)v.y;
    o.h[2] = (_Float16)v.z; o.h[3] = (_Float16)v.w;
    *(uint2*)(y + (size_t)i * 4) = o.u;
  }
}

// ---------------------------------------------------------------------------
// LayerNorm over last dim (1024), f32 in -> f16 out
// ---------------------------------------------------------------------------
__global__ __launch_bounds__(256)
void ln_kernel(const float* __restrict__ x, const float* __restrict__ g,
               const float* __restrict__ bta, _Float16* __restrict__ y) {
  __shared__ float rs[256], rs2[256];
  __shared__ float s_mu, s_rstd;
  const int t = threadIdx.x;
  const size_t base = (size_t)blockIdx.x * 1024;
  float v[4];
  float s = 0.f, s2 = 0.f;
#pragma unroll
  for (int i = 0; i < 4; ++i) {
    v[i] = x[base + t + i * 256];
    s += v[i]; s2 += v[i] * v[i];
  }
  rs[t] = s; rs2[t] = s2;
  __syncthreads();
  for (int o = 128; o > 0; o >>= 1) {
    if (t < o) { rs[t] += rs[t + o]; rs2[t] += rs2[t + o]; }
    __syncthreads();
  }
  if (t == 0) {
    float mu = rs[0] * (1.0f / 1024.0f);
    float var = rs2[0] * (1.0f / 1024.0f) - mu * mu;
    s_mu = mu;
    s_rstd = rsqrtf(var + 1e-5f);
  }
  __syncthreads();
#pragma unroll
  for (int i = 0; i < 4; ++i) {
    int c = t + i * 256;
    y[base + c] = (_Float16)((v[i] - s_mu) * s_rstd * g[c] + bta[c]);
  }
}

// ---------------------------------------------------------------------------
// Tiled WMMA GEMM, double-buffered LDS ping-pong:
//   C[M,N] = act( (A[M,K] @ B[K,N] + bias) * scale [+ residual] )
// 256 threads = 8 waves; block tile 128x128, K-step 32; wave tile 64x32.
// One barrier per K-step; next tile's global loads overlap current WMMAs.
// ---------------------------------------------------------------------------
template <bool DO_GELU, bool HAS_RES, bool OUT_F16>
__global__ __launch_bounds__(256)
void gemm_f16_kernel(const _Float16* __restrict__ A, const _Float16* __restrict__ Bw,
                     const float* __restrict__ bias, const float* __restrict__ residual,
                     float* __restrict__ outF32, _Float16* __restrict__ outF16,
                     int N, int K, float scale) {
  __shared__ _Float16 sA[2][128 * 40];   // [m][k], stride 40 halves
  __shared__ _Float16 sB[2][128 * 40];   // B^T tile: [n][k], stride 40 halves

  const int t = threadIdx.x;
  const int lane = t & 31;
  const int wave = t >> 5;
  const int wr = wave >> 2;              // 0..1
  const int wc = wave & 3;               // 0..3
  const int bn = blockIdx.x * 128;
  const int bm = blockIdx.y * 128;

  // per-thread staging coordinates
  const int rA0 = t >> 2;                 // A: rows, 2 chunks per thread
  const int cA0 = (t & 3) << 3;
  const int kkB = t >> 3;                 // B: one k-row, 16 cols per thread
  const int ncB = (t & 7) << 4;

  uint4 ra0, ra1, rb0, rb1;
  auto gload = [&](int k0) {
    ra0 = *(const uint4*)(A + (size_t)(bm + rA0) * K + k0 + cA0);
    ra1 = *(const uint4*)(A + (size_t)(bm + rA0 + 64) * K + k0 + cA0);
    const _Float16* bp = Bw + (size_t)(k0 + kkB) * N + bn + ncB;
    rb0 = *(const uint4*)bp;
    rb1 = *(const uint4*)(bp + 8);
  };
  auto stage = [&](int buf) {
    *(uint4*)(&sA[buf][rA0 * 40 + cA0]) = ra0;
    *(uint4*)(&sA[buf][(rA0 + 64) * 40 + cA0]) = ra1;
    H8 d0, d1;
    d0.q = rb0; d1.q = rb1;
#pragma unroll
    for (int j = 0; j < 8; ++j) {
      sB[buf][(ncB + j) * 40 + kkB]     = d0.h[j];
      sB[buf][(ncB + 8 + j) * 40 + kkB] = d1.h[j];
    }
  };

  const v8f vzero = {0.f, 0.f, 0.f, 0.f, 0.f, 0.f, 0.f, 0.f};
  v8f acc[4][2];
#pragma unroll
  for (int i = 0; i < 4; ++i)
#pragma unroll
    for (int j = 0; j < 2; ++j) acc[i][j] = vzero;

  gload(0);
  stage(0);
  int cur = 0;

  for (int k0 = 0; k0 < K; k0 += 32) {
    const bool has_next = (k0 + 32) < K;
    if (has_next) gload(k0 + 32);            // in flight during compute
    if (k0 + 64 < K) {                       // L2 prefetch two tiles ahead
      __builtin_prefetch(A + (size_t)(bm + rA0) * K + k0 + 64 + cA0, 0, 1);
      __builtin_prefetch(Bw + (size_t)(k0 + 64 + kkB) * N + bn + ncB, 0, 1);
    }
    __syncthreads();                         // buf[cur] fully staged

    v16h af[4], bf[2];
#pragma unroll
    for (int i = 0; i < 4; ++i)
      af[i] = load_frag_a(&sA[cur][(wr * 64 + i * 16) * 40], 40, lane);
#pragma unroll
    for (int j = 0; j < 2; ++j)
      bf[j] = load_frag_b(&sB[cur][(wc * 32 + j * 16) * 40], 40, lane);
#pragma unroll
    for (int i = 0; i < 4; ++i)
#pragma unroll
      for (int j = 0; j < 2; ++j)
        acc[i][j] = __builtin_amdgcn_wmma_f32_16x16x32_f16(
            false, af[i], false, bf[j], (short)0, acc[i][j], false, false);

    if (has_next) stage(cur ^ 1);            // write other buffer; readers of it
    cur ^= 1;                                // were ordered by this iter's barrier
  }

  // epilogue
  const int rbase = bm + wr * 64 + ((lane >> 4) << 3);
  const int cbase = bn + wc * 32 + (lane & 15);
#pragma unroll
  for (int i = 0; i < 4; ++i) {
#pragma unroll
    for (int j = 0; j < 2; ++j) {
      const int col = cbase + j * 16;
      const float bs = bias[col];
#pragma unroll
      for (int e = 0; e < 8; ++e) {
        const int row = rbase + i * 16 + e;
        float val = (acc[i][j][e] + bs) * scale;
        if constexpr (HAS_RES) val += residual[(size_t)row * N + col];
        if constexpr (DO_GELU) val = 0.5f * val * (1.0f + erff(val * 0.70710678118f));
        if constexpr (OUT_F16) outF16[(size_t)row * N + col] = (_Float16)val;
        else                   outF32[(size_t)row * N + col] = val;
      }
    }
  }
}

// ---------------------------------------------------------------------------
// Flash attention: one block = 64 query rows of one (b,h). 128 threads = 4 waves,
// each wave owns 16 query rows. Streams Nk=2048 keys in chunks of 64 with
// online softmax; both einsums via WMMA f16->f32.
// q is pre-scaled by 1/sqrt(head_dim) in the Q projection.
// ---------------------------------------------------------------------------
__global__ __launch_bounds__(128)
void attn_kernel(const _Float16* __restrict__ Qh, const _Float16* __restrict__ Kh,
                 const _Float16* __restrict__ Vh, _Float16* __restrict__ ctx) {
  __shared__ _Float16 sQ[64 * 72];    // [q][dim]
  __shared__ _Float16 sK[64 * 72];    // [key][dim]  (== B^T for Q@K^T)
  __shared__ _Float16 sVt[64 * 72];   // [dim][key]  (== B^T for P@V)
  __shared__ float    sS[64 * 64];
  __shared__ _Float16 sP[64 * 72];    // [q][key]
  __shared__ float    sAlpha[64];
  __shared__ float    sL[64];

  const int t = threadIdx.x;
  const int lane = t & 31;
  const int wave = t >> 5;            // 0..3
  const int b = blockIdx.y >> 4;
  const int h = blockIdx.y & 15;
  const int q0 = blockIdx.x * 64;
  const size_t qrow0 = (size_t)b * 1024 + q0;
  const size_t krow0 = (size_t)b * 2048;
  const int hc = h * 64;

  // load Q tile (64x64 halves)
#pragma unroll
  for (int i = 0; i < 4; ++i) {
    int c = t + i * 128;
    int r = c >> 3;
    int co = (c & 7) << 3;
    uint4 dq = *(const uint4*)(Qh + (qrow0 + r) * 1024 + hc + co);
    *(uint4*)(sQ + r * 72 + co) = dq;
  }

  float m_run = -3.0e38f, l_run = 0.0f;
  const v8f vzero = {0.f, 0.f, 0.f, 0.f, 0.f, 0.f, 0.f, 0.f};
  v8f o[4];
#pragma unroll
  for (int dd = 0; dd < 4; ++dd) o[dd] = vzero;

  const int rl = ((lane >> 4) << 3);

  for (int kc = 0; kc < 2048; kc += 64) {
    __syncthreads();
    // load K chunk and V chunk (transposed) into LDS
#pragma unroll
    for (int i = 0; i < 4; ++i) {
      int c = t + i * 128;
      int r = c >> 3;
      int co = (c & 7) << 3;
      uint4 dk = *(const uint4*)(Kh + (krow0 + kc + r) * 1024 + hc + co);
      *(uint4*)(sK + r * 72 + co) = dk;
      H8 dv;
      dv.q = *(const uint4*)(Vh + (krow0 + kc + r) * 1024 + hc + co);
#pragma unroll
      for (int j = 0; j < 8; ++j) sVt[(co + j) * 72 + r] = dv.h[j];
    }
    __syncthreads();

    // S = Q @ K^T for this wave's 16 query rows x 64 keys
    v8f s[4];
#pragma unroll
    for (int j = 0; j < 4; ++j) s[j] = vzero;
#pragma unroll
    for (int ks = 0; ks < 64; ks += 32) {
      v16h a = load_frag_a(sQ + (wave * 16) * 72 + ks, 72, lane);
#pragma unroll
      for (int j = 0; j < 4; ++j) {
        v16h bfr = load_frag_b(sK + (j * 16) * 72 + ks, 72, lane);
        s[j] = __builtin_amdgcn_wmma_f32_16x16x32_f16(false, a, false, bfr,
                                                      (short)0, s[j], false, false);
      }
    }
#pragma unroll
    for (int j = 0; j < 4; ++j)
#pragma unroll
      for (int e = 0; e < 8; ++e)
        sS[(wave * 16 + rl + e) * 64 + j * 16 + (lane & 15)] = s[j][e];
    __syncthreads();

    // online softmax: lanes 0..15 each own one query row of the wave's tile
    if (lane < 16) {
      const int row = wave * 16 + lane;
      const float* sr = sS + row * 64;
      float cm = sr[0];
#pragma unroll 8
      for (int j = 1; j < 64; ++j) cm = fmaxf(cm, sr[j]);
      float nm = fmaxf(m_run, cm);
      float alpha = __expf(m_run - nm);
      float ps = 0.0f;
#pragma unroll 8
      for (int j = 0; j < 64; ++j) {
        float p = __expf(sr[j] - nm);
        ps += p;
        sP[row * 72 + j] = (_Float16)p;
      }
      l_run = l_run * alpha + ps;
      m_run = nm;
      sAlpha[row] = alpha;
    }
    __syncthreads();

    // rescale running O accumulators by alpha[row]
#pragma unroll
    for (int dd = 0; dd < 4; ++dd)
#pragma unroll
      for (int e = 0; e < 8; ++e)
        o[dd][e] *= sAlpha[wave * 16 + rl + e];

    // O += P @ V
#pragma unroll
    for (int ks = 0; ks < 64; ks += 32) {
      v16h a = load_frag_a(sP + (wave * 16) * 72 + ks, 72, lane);
#pragma unroll
      for (int dd = 0; dd < 4; ++dd) {
        v16h bfr = load_frag_b(sVt + (dd * 16) * 72 + ks, 72, lane);
        o[dd] = __builtin_amdgcn_wmma_f32_16x16x32_f16(false, a, false, bfr,
                                                       (short)0, o[dd], false, false);
      }
    }
  }

  if (lane < 16) sL[wave * 16 + lane] = l_run;
  __syncthreads();

#pragma unroll
  for (int dd = 0; dd < 4; ++dd)
#pragma unroll
    for (int e = 0; e < 8; ++e) {
      const int row = wave * 16 + rl + e;
      float val = o[dd][e] / sL[row];
      ctx[(qrow0 + row) * 1024 + hc + dd * 16 + (lane & 15)] = (_Float16)val;
    }
}

// ---------------------------------------------------------------------------
// Host orchestration
// ---------------------------------------------------------------------------
extern "C" void kernel_launch(void* const* d_in, const int* in_sizes, int n_in,
                              void* d_out, int out_size, void* d_ws, size_t ws_size,
                              hipStream_t stream) {
  (void)in_sizes; (void)n_in; (void)out_size; (void)ws_size;

  const float* tgt  = (const float*)d_in[0];
  const float* emb  = (const float*)d_in[1];
  const float* ln_g = (const float*)d_in[2];
  const float* ln_b = (const float*)d_in[3];
  const float* wq = (const float*)d_in[4];   const float* bq = (const float*)d_in[5];
  const float* wk = (const float*)d_in[6];   const float* bk = (const float*)d_in[7];
  const float* wv = (const float*)d_in[8];   const float* bv = (const float*)d_in[9];
  const float* wo = (const float*)d_in[10];  const float* bo = (const float*)d_in[11];
  const float* w1 = (const float*)d_in[12];  const float* b1 = (const float*)d_in[13];
  const float* w2 = (const float*)d_in[14];  const float* b2 = (const float*)d_in[15];

  char* ws = (char*)d_ws;
  size_t off = 0;
  auto alloc = [&](size_t bytes) -> void* {
    void* p = ws + off;
    off += (bytes + 255) & ~(size_t)255;
    return p;
  };
  _Float16* emb16 = (_Float16*)alloc(8192ull * 1024 * 2);
  _Float16* qn16  = (_Float16*)alloc(4096ull * 1024 * 2);
  _Float16* q16   = (_Float16*)alloc(4096ull * 1024 * 2);
  _Float16* k16   = (_Float16*)alloc(8192ull * 1024 * 2);
  _Float16* v16   = (_Float16*)alloc(8192ull * 1024 * 2);
  _Float16* ctx16 = (_Float16*)alloc(4096ull * 1024 * 2);
  float*    tgt2  = (float*)   alloc(4096ull * 1024 * 4);
  _Float16* ln216 = (_Float16*)alloc(4096ull * 1024 * 2);
  _Float16* h16   = (_Float16*)alloc(4096ull * 4096 * 2);
  _Float16* wq16  = (_Float16*)alloc(1024ull * 1024 * 2);
  _Float16* wk16  = (_Float16*)alloc(1024ull * 1024 * 2);
  _Float16* wv16  = (_Float16*)alloc(1024ull * 1024 * 2);
  _Float16* wo16  = (_Float16*)alloc(1024ull * 1024 * 2);
  _Float16* w116  = (_Float16*)alloc(1024ull * 4096 * 2);
  _Float16* w216  = (_Float16*)alloc(4096ull * 1024 * 2);

  auto cvt = [&](const float* src, _Float16* dst, size_t n) {
    int n4 = (int)(n / 4);
    cvt_kernel<<<(n4 + 255) / 256, 256, 0, stream>>>(src, dst, n4);
  };
  cvt(emb, emb16, 8192ull * 1024);
  cvt(wq, wq16, 1024ull * 1024);
  cvt(wk, wk16, 1024ull * 1024);
  cvt(wv, wv16, 1024ull * 1024);
  cvt(wo, wo16, 1024ull * 1024);
  cvt(w1, w116, 1024ull * 4096);
  cvt(w2, w216, 4096ull * 1024);

  // q-norm (ln of tgt)
  ln_kernel<<<4096, 256, 0, stream>>>(tgt, ln_g, ln_b, qn16);

  dim3 blk(256);
  // Q projection: scale folds 1/sqrt(head_dim)=1/8 into the output
  gemm_f16_kernel<false, false, true><<<dim3(1024 / 128, 4096 / 128), blk, 0, stream>>>(
      qn16, wq16, bq, nullptr, nullptr, q16, 1024, 1024, 0.125f);
  gemm_f16_kernel<false, false, true><<<dim3(1024 / 128, 8192 / 128), blk, 0, stream>>>(
      emb16, wk16, bk, nullptr, nullptr, k16, 1024, 1024, 1.0f);
  gemm_f16_kernel<false, false, true><<<dim3(1024 / 128, 8192 / 128), blk, 0, stream>>>(
      emb16, wv16, bv, nullptr, nullptr, v16, 1024, 1024, 1.0f);

  // attention: grid (Nq/64, B*H)
  attn_kernel<<<dim3(16, 64), 128, 0, stream>>>(q16, k16, v16, ctx16);

  // O projection + residual (fp32 out)
  gemm_f16_kernel<false, true, false><<<dim3(1024 / 128, 4096 / 128), blk, 0, stream>>>(
      ctx16, wo16, bo, tgt, tgt2, nullptr, 1024, 1024, 1.0f);

  // MLP block (same LN params per reference)
  ln_kernel<<<4096, 256, 0, stream>>>(tgt2, ln_g, ln_b, ln216);
  gemm_f16_kernel<true, false, true><<<dim3(4096 / 128, 4096 / 128), blk, 0, stream>>>(
      ln216, w116, b1, nullptr, nullptr, h16, 4096, 1024, 1.0f);
  gemm_f16_kernel<false, true, false><<<dim3(1024 / 128, 4096 / 128), blk, 0, stream>>>(
      h16, w216, b2, tgt2, (float*)d_out, nullptr, 1024, 4096, 1.0f);
}
